// ShiftedWindowAttention_15255723835599
// MI455X (gfx1250) — matrix operand order repaired and checked by reading, not compiled, not run on hardware
//
#include <hip/hip_runtime.h>

typedef __attribute__((ext_vector_type(16))) __bf16 v16bf;
typedef __attribute__((ext_vector_type(8)))  float  v8f;

union Frag {
  v16bf v;
  __bf16 h[16];
  uint4  q[2];
};

__device__ __forceinline__ v8f wmma_bf16(v16bf a, v16bf b, v8f c) {
  // D = A(16x32 bf16) * B(32x16 bf16) + C(16x16 f32)
  return __builtin_amdgcn_wmma_f32_16x16x32_bf16(false, a, false, b, (short)0, c,
                                                 false, false);
}

// Async global->LDS 16B copy (gfx1250 GLOBAL_LOAD_ASYNC_TO_LDS_B128, ASYNCcnt).
// Flat addresses to LDS carry the LDS byte offset in addr[31:0] (ISA 10.2).
__device__ __forceinline__ void async_copy_b128(__bf16* lds_dst, const __bf16* g_src) {
  const uint32_t lds = (uint32_t)(uintptr_t)lds_dst;
  asm volatile("global_load_async_to_lds_b128 %0, %1, off"
               :: "v"(lds), "v"(g_src)
               : "memory");
}

// A fragment 16x32 (MxK) from row-major LDS [row][col].
// lanes 0-15: M=lane, K runs {k0..k0+7, k0+16..k0+23}; lanes 16-31: K runs +8.
__device__ __forceinline__ v16bf load_frag_a(const __bf16* p, int stride, int m0,
                                             int k0, int lane) {
  const __bf16* row = p + (m0 + (lane & 15)) * stride;
  const int kb = k0 + ((lane >> 4) << 3);
  Frag f;
  f.q[0] = *reinterpret_cast<const uint4*>(row + kb);
  f.q[1] = *reinterpret_cast<const uint4*>(row + kb + 16);
  return f.v;
}

// B fragment 32x16 (KxN) where LDS holds B transposed: src[n][k].
// N = lane&15; lanes 0-15 K runs {k0..k0+7, k0+16..k0+23}; lanes 16-31: K runs +8.
__device__ __forceinline__ v16bf load_frag_bT(const __bf16* p, int stride, int n0,
                                              int k0, int lane) {
  const __bf16* row = p + (n0 + (lane & 15)) * stride;
  const int kb = k0 + ((lane >> 4) << 3);
  Frag f;
  f.q[0] = *reinterpret_cast<const uint4*>(row + kb);
  f.q[1] = *reinterpret_cast<const uint4*>(row + kb + 16);
  return f.v;
}

constexpr int C_   = 256;
constexpr int H_   = 128;
constexpr int W2_  = 128;
constexpr int SEQ_ = 64;
constexpr int HD_  = 8;

constexpr int QK_LD = 520;  // 512 + 8 pad (1040 B row, 16B aligned)
constexpr int VT_LD = 72;   // 64  + 8 pad
constexpr int A_LD  = 264;  // 256 + 8 pad
constexpr int BT_LD = 40;   // 32  + 8 pad
constexpr int P_LD  = 72;   // 64  + 8 pad
constexpr int SCR_ELEMS = SEQ_ * A_LD + 3 * 64 * BT_LD;  // 24576 elems (>= P: 9216)

// ---------------------------------------------------------------------------
// Prep: transpose + convert fp32 [K][N] -> bf16 [N][K] (for async bf16 staging)
// ---------------------------------------------------------------------------
__global__ __launch_bounds__(256) void transpose_cvt(const float* __restrict__ src,
                                                     __bf16* __restrict__ dst,
                                                     int K, int N) {
  __shared__ float tile[32][33];
  const int nb = N >> 5;
  const int bx = blockIdx.x % nb;
  const int by = blockIdx.x / nb;
  const int tx = threadIdx.x & 31;
  const int ty = threadIdx.x >> 5;
#pragma unroll
  for (int j = 0; j < 32; j += 8)
    tile[ty + j][tx] = src[(size_t)(by * 32 + ty + j) * N + bx * 32 + tx];
  __syncthreads();
#pragma unroll
  for (int j = 0; j < 32; j += 8)
    dst[(size_t)(bx * 32 + ty + j) * K + by * 32 + tx] = (__bf16)tile[tx][ty + j];
}

// ---------------------------------------------------------------------------
// Kernel A: per (batch, window) block, 8 waves.
//   phase 1: qkv[64x768] = bf16(x_win[64x256] @ w_qkv + b_qkv) via WMMA.
//            q,k parked row-major in LDS; v parked TRANSPOSED (s_vT[c][t]).
//            Weight tiles streamed with async-to-LDS 3-buffer pipeline.
//   phase 2: wave == head: S = q k^T * scale + bias; softmax; O = P v (WMMA).
// ---------------------------------------------------------------------------
__global__ __launch_bounds__(256) void swin_qkv_attn(
    const float* __restrict__ x, const float* __restrict__ b_qkv,
    const float* __restrict__ bias_table, const __bf16* __restrict__ wqkvT,
    __bf16* __restrict__ attn_out) {
  __shared__ __align__(16) __bf16 s_qk[SEQ_ * QK_LD];      // 66,560 B
  __shared__ __align__(16) __bf16 s_vT[256 * VT_LD];       // 36,864 B (vc = head*32+c)
  __shared__ __align__(16) float  s_bias[225 * HD_];       //  7,200 B
  __shared__ __align__(16) __bf16 s_scr[SCR_ELEMS];        // 49,152 B

  __bf16* s_A  = s_scr;                 // 64 x A_LD (phase 1 only)
  __bf16* s_BT = s_scr + SEQ_ * A_LD;   // 3 x (64 x BT_LD) async ring

  const int tid  = threadIdx.x;
  const int wave = tid >> 5;
  const int lane = tid & 31;
  const int bat  = blockIdx.x >> 8;
  const int win  = blockIdx.x & 255;
  const int hw   = win >> 4;
  const int ww   = win & 15;

  // issue async weight tile t: 64 rows(n) x 32 k bf16 from wqkvT[n][k]
  auto issue_tile = [&](int t) {
    const int tn0 = (t >> 3) * 64;
    const int tk0 = (t & 7) * 32;
    __bf16* buf = s_BT + (t % 3) * (64 * BT_LD);
    const int nn = tid >> 2;
    const int ch = tid & 3;
    async_copy_b128(buf + nn * BT_LD + ch * 8,
                    wqkvT + (size_t)(tn0 + nn) * 256 + tk0 + ch * 8);
  };

  for (int i = tid; i < 225 * HD_; i += 256) s_bias[i] = bias_table[i];

  // Gather shifted-window tile: A[row][c] = x[b][c][(hw*8+wy+4)%128][(ww*8+wx+4)%128]
  {
    const int row = tid & 63;
    const int c0  = tid >> 6;  // 0..3
    const int wy = row >> 3, wx = row & 7;
    const int hs  = (hw * 8 + wy + 4) & 127;
    const int wss = (ww * 8 + wx + 4) & 127;
    const float* xp = x + ((size_t)bat * C_) * (H_ * W2_) + hs * W2_ + wss;
    __bf16* ar = s_A + row * A_LD;
#pragma unroll 4
    for (int i = 0; i < 64; ++i) {
      const int c = c0 + (i << 2);
      ar[c] = (__bf16)xp[(size_t)c * (H_ * W2_)];
    }
  }
  issue_tile(0);
  issue_tile(1);
  __syncthreads();

  const int msub  = wave >> 1;         // 4 M-subtiles of the 64-row window
  const int nhalf = (wave & 1) << 1;   // which pair of N-subtiles in a 64 chunk
  const int rlo   = (lane >> 4) << 3;  // C-layout: upper lanes hold M+8
  const int ncol  = lane & 15;

  // Hoist all 8 A fragments (full K=256); s_A is dead afterwards.
  v16bf afrag[8];
#pragma unroll
  for (int kc = 0; kc < 8; ++kc)
    afrag[kc] = load_frag_a(s_A, A_LD, msub * 16, kc * 32, lane);

  // ---- phase 1: QKV GEMM, 96 weight tiles (12 N-chunks x 8 K-chunks) ----
  for (int nc = 0; nc < 12; ++nc) {
    const int n0 = nc * 64;
    v8f acc0 = (v8f){0.f, 0.f, 0.f, 0.f, 0.f, 0.f, 0.f, 0.f};
    v8f acc1 = acc0;
    for (int kc = 0; kc < 8; ++kc) {
      const int t = nc * 8 + kc;
      if (t < 95) {  // drain tile t, keep t+1 in flight
        asm volatile("s_wait_asynccnt 0x1" ::: "memory");
      } else {
        asm volatile("s_wait_asynccnt 0x0" ::: "memory");
      }
      __syncthreads();  // tile t visible to all; separates reads(t-1) from issue(t+2)
      const __bf16* buf = s_BT + (t % 3) * (64 * BT_LD);
      const v16bf fb0 = load_frag_bT(buf, BT_LD, nhalf * 16, 0, lane);
      const v16bf fb1 = load_frag_bT(buf, BT_LD, (nhalf + 1) * 16, 0, lane);
      acc0 = wmma_bf16(afrag[kc], fb0, acc0);
      acc1 = wmma_bf16(afrag[kc], fb1, acc1);
      if (t + 2 < 96) issue_tile(t + 2);
    }
    // epilogue: + bias, convert; q,k row-major, v transposed
    const int col0  = n0 + nhalf * 16 + ncol;
    const float bb0 = b_qkv[col0];
    const float bb1 = b_qkv[col0 + 16];
    if (n0 < 512) {
#pragma unroll
      for (int r = 0; r < 8; ++r) {
        const int m = msub * 16 + rlo + r;
        s_qk[m * QK_LD + col0]      = (__bf16)(acc0[r] + bb0);
        s_qk[m * QK_LD + col0 + 16] = (__bf16)(acc1[r] + bb1);
      }
    } else {
      const int vc0 = col0 - 512;  // vc = head*32 + c
#pragma unroll
      for (int r = 0; r < 8; ++r) {
        const int m = msub * 16 + rlo + r;
        s_vT[vc0 * VT_LD + m]        = (__bf16)(acc0[r] + bb0);
        s_vT[(vc0 + 16) * VT_LD + m] = (__bf16)(acc1[r] + bb1);
      }
    }
  }
  __syncthreads();

  // ---- phase 2: attention, wave == head ----
  const int head = wave;
  const int qoff = head * 32;
  const int koff = 256 + head * 32;
  const __bf16* vt = s_vT + head * 32 * VT_LD;  // [c][t]
  __bf16* sP = s_scr + wave * (16 * P_LD);      // per-wave 16x64 P strip
  const float scale = 0.17677669529663687f;     // 1/sqrt(32)

  for (int ms = 0; ms < 4; ++ms) {
    v8f sc[4];
#pragma unroll
    for (int ns = 0; ns < 4; ++ns) sc[ns] = (v8f){0.f, 0.f, 0.f, 0.f, 0.f, 0.f, 0.f, 0.f};
    const v16bf aq = load_frag_a(s_qk, QK_LD, ms * 16, qoff, lane);
#pragma unroll
    for (int ns = 0; ns < 4; ++ns) {
      const v16bf bk = load_frag_bT(s_qk, QK_LD, ns * 16, koff, lane);
      sc[ns] = wmma_bf16(aq, bk, sc[ns]);
    }
    // scale + relative-position bias + row softmax (rows live in 16-lane halves)
#pragma unroll
    for (int r = 0; r < 8; ++r) {
      const int m = ms * 16 + rlo + r;
      const int ys = m >> 3, xs = m & 7;
      float mx = -3.0e38f;
#pragma unroll
      for (int ns = 0; ns < 4; ++ns) {
        const int t = ns * 16 + ncol;
        const int rel = (ys - (t >> 3) + 7) * 15 + (xs - (t & 7) + 7);
        const float v = sc[ns][r] * scale + s_bias[rel * HD_ + head];
        sc[ns][r] = v;
        mx = fmaxf(mx, v);
      }
#pragma unroll
      for (int off = 8; off >= 1; off >>= 1) mx = fmaxf(mx, __shfl_xor(mx, off, 32));
      float sum = 0.f;
#pragma unroll
      for (int ns = 0; ns < 4; ++ns) {
        const float e = __expf(sc[ns][r] - mx);
        sc[ns][r] = e;
        sum += e;
      }
#pragma unroll
      for (int off = 8; off >= 1; off >>= 1) sum += __shfl_xor(sum, off, 32);
      const float inv = 1.0f / sum;
#pragma unroll
      for (int ns = 0; ns < 4; ++ns)
        sP[(rlo + r) * P_LD + ns * 16 + ncol] = (__bf16)(sc[ns][r] * inv);
    }
    // same-wave LDS is in-order (DScnt); fence compiler + wait before P reload
    asm volatile("s_wait_dscnt 0x0" ::: "memory");

    // O strip = P(16x64) @ V(64x32): K in 2 chunks; V via transposed LDS copy
    v8f o0 = (v8f){0.f, 0.f, 0.f, 0.f, 0.f, 0.f, 0.f, 0.f};
    v8f o1 = o0;
#pragma unroll
    for (int kc = 0; kc < 2; ++kc) {
      const v16bf ap  = load_frag_a(sP, P_LD, 0, kc * 32, lane);
      const v16bf bv0 = load_frag_bT(vt, VT_LD, 0, kc * 32, lane);
      const v16bf bv1 = load_frag_bT(vt, VT_LD, 16, kc * 32, lane);
      o0 = wmma_bf16(ap, bv0, o0);
      o1 = wmma_bf16(ap, bv1, o1);
    }
    // store 16 rows, un-windowed + inverse roll (+SHIFT), merged-head column
#pragma unroll
    for (int r = 0; r < 8; ++r) {
      const int m = ms * 16 + rlo + r;
      const int wy = m >> 3, wx = m & 7;
      const int ho = (hw * 8 + wy + 4) & 127;
      const int wo = (ww * 8 + wx + 4) & 127;
      __bf16* op =
          attn_out + (((size_t)bat * H_ + ho) * W2_ + wo) * C_ + head * 32 + ncol;
      op[0]  = (__bf16)o0[r];
      op[16] = (__bf16)o1[r];
    }
    asm volatile("s_wait_dscnt 0x0" ::: "memory");  // sP strip reuse across ms
  }
}

// ---------------------------------------------------------------------------
// Kernel B: output projection [65536,256]bf16 @ [256,256] + b_out -> NCHW fp32
// M tile 128 (one (b,h) strip of w), N tile 64. Async 3-buffer staging of both
// the activation tile and the transposed bf16 weight tile; LDS-transposed store.
// ---------------------------------------------------------------------------
__global__ __launch_bounds__(256) void swin_proj(
    const __bf16* __restrict__ ain, const __bf16* __restrict__ woutT,
    const float* __restrict__ b_out, float* __restrict__ out) {
  __shared__ __align__(16) __bf16 sA[3 * 128 * 40];   // 30,720 B
  __shared__ __align__(16) __bf16 sBT[3 * 64 * 40];   // 15,360 B
  __shared__ __align__(16) float  sC[128 * 65];       // 33,280 B

  const int tid  = threadIdx.x;
  const int wave = tid >> 5;
  const int lane = tid & 31;
  const int rlo  = (lane >> 4) << 3;
  const int ncol = lane & 15;
  const size_t m0 = (size_t)(blockIdx.x >> 2) * 128;
  const int n0   = (blockIdx.x & 3) * 64;

  auto issue_tile = [&](int kc) {
    const int k0 = kc * 32;
    __bf16* bufA = sA + (kc % 3) * (128 * 40);
    __bf16* bufB = sBT + (kc % 3) * (64 * 40);
#pragma unroll
    for (int j = 0; j < 2; ++j) {  // A: 128 rows x 64B = 512 chunks of 16B
      const int id = tid + j * 256;
      const int row = id >> 2, c4 = id & 3;
      async_copy_b128(bufA + row * 40 + c4 * 8,
                      ain + (m0 + row) * 256 + k0 + c4 * 8);
    }
    const int nn = tid >> 2;       // W^T: 64 rows x 64B = 256 chunks
    async_copy_b128(bufB + nn * 40 + (tid & 3) * 8,
                    woutT + (size_t)(n0 + nn) * 256 + k0 + (tid & 3) * 8);
  };

  v8f acc[4];
#pragma unroll
  for (int ns = 0; ns < 4; ++ns) acc[ns] = (v8f){0.f, 0.f, 0.f, 0.f, 0.f, 0.f, 0.f, 0.f};

  issue_tile(0);
  issue_tile(1);
  for (int kc = 0; kc < 8; ++kc) {
    if (kc < 7) {  // drain tile kc (3 async ops), keep kc+1 flying
      asm volatile("s_wait_asynccnt 0x3" ::: "memory");
    } else {
      asm volatile("s_wait_asynccnt 0x0" ::: "memory");
    }
    __syncthreads();
    const __bf16* bufA = sA + (kc % 3) * (128 * 40);
    const __bf16* bufB = sBT + (kc % 3) * (64 * 40);
    const v16bf a = load_frag_a(bufA, 40, wave * 16, 0, lane);
#pragma unroll
    for (int ns = 0; ns < 4; ++ns) {
      const v16bf fb = load_frag_bT(bufB, 40, ns * 16, 0, lane);
      acc[ns] = wmma_bf16(a, fb, acc[ns]);
    }
    if (kc + 2 < 8) issue_tile(kc + 2);
  }
  // bias + stage fp32 tile for transposed (coalesced NCHW) store
#pragma unroll
  for (int ns = 0; ns < 4; ++ns) {
    const int n  = ns * 16 + ncol;
    const float bb = b_out[n0 + n];
#pragma unroll
    for (int r = 0; r < 8; ++r) sC[(wave * 16 + rlo + r) * 65 + n] = acc[ns][r] + bb;
  }
  __syncthreads();
  // out[b][n0+n][h][w] — 512 B contiguous runs along w
  const int hh = (int)((m0 >> 7) & 127);
  const int bb = (int)(m0 >> 14);
#pragma unroll 4
  for (int j = 0; j < 32; ++j) {
    const int idx = tid + j * 256;
    const int n = idx >> 7;
    const int m = idx & 127;
    out[(((size_t)bb * 256 + n0 + n) * 128 + hh) * 128 + m] = sC[m * 65 + n];
  }
}

extern "C" void kernel_launch(void* const* d_in, const int* in_sizes, int n_in,
                              void* d_out, int out_size, void* d_ws, size_t ws_size,
                              hipStream_t stream) {
  const float* x        = (const float*)d_in[0];
  const float* w_qkv    = (const float*)d_in[1];
  const float* b_qkv    = (const float*)d_in[2];
  const float* w_out    = (const float*)d_in[3];
  const float* b_out    = (const float*)d_in[4];
  const float* bias_tab = (const float*)d_in[5];
  float* out = (float*)d_out;

  char* ws = (char*)d_ws;
  __bf16* attn  = (__bf16*)ws;                          // 65536 x 256 bf16 = 32 MB
  __bf16* wqkvT = (__bf16*)(ws + (size_t)33554432);     // 768 x 256 bf16
  __bf16* woutT = (__bf16*)(ws + (size_t)33554432 + 393216);  // 256 x 256 bf16

  transpose_cvt<<<dim3(8 * 24), 256, 0, stream>>>(w_qkv, wqkvT, 256, 768);
  transpose_cvt<<<dim3(8 * 8), 256, 0, stream>>>(w_out, woutT, 256, 256);
  swin_qkv_attn<<<dim3(1024), 256, 0, stream>>>(x, b_qkv, bias_tab, wqkvT, attn);
  swin_proj<<<dim3(2048), 256, 0, stream>>>(attn, woutT, b_out, out);
}
